// Model_39676907880769
// MI455X (gfx1250) — matrix operand order repaired and checked by reading, not compiled, not run on hardware
//
#include <hip/hip_runtime.h>

// ---- CDNA5 wave32 WMMA types -------------------------------------------------
typedef __attribute__((ext_vector_type(16))) __bf16 v16bf;
typedef __attribute__((ext_vector_type(8)))  __bf16 v8bf;
typedef __attribute__((ext_vector_type(8)))  float  v8f;

union V16U { v16bf v; v8bf h[2]; };
union V8U  { v8bf  v; __bf16 e[8]; };

#define SEQ   4096
#define DHEAD 64
#define WAVES 8
#define ROWS_PER_WAVE 16
#define ROWS_PER_BLOCK (WAVES * ROWS_PER_WAVE)   // 128
#define CHUNK 64                                  // kv rows per iteration

// Q prescale: scores*1024, and move exp() into exp2() domain: 1024*log2(e)
#define QSCALE 1477.3197218702985f

// ---- 16-lane butterfly reductions in pure VALU via DPP16 ---------------------
// masks stay within each 16-lane half of the wave (matches WMMA C/D row halves)
template <int CTRL>
__device__ __forceinline__ float dpp_mov(float x) {
    return __int_as_float(__builtin_amdgcn_update_dpp(
        __float_as_int(x), __float_as_int(x), CTRL, 0xF, 0xF, true));
}
__device__ __forceinline__ float redmax16(float v) {
    v = fmaxf(v, dpp_mov<0xB1>(v));   // quad_perm [1,0,3,2]  (xor 1)
    v = fmaxf(v, dpp_mov<0x4E>(v));   // quad_perm [2,3,0,1]  (xor 2)
    v = fmaxf(v, dpp_mov<0x141>(v));  // row_half_mirror      (pairs across 4-groups)
    v = fmaxf(v, dpp_mov<0x140>(v));  // row_mirror           (pairs across 8-groups)
    return v;
}
__device__ __forceinline__ float redsum16(float v) {
    v += dpp_mov<0xB1>(v);
    v += dpp_mov<0x4E>(v);
    v += dpp_mov<0x141>(v);
    v += dpp_mov<0x140>(v);
    return v;
}
__device__ __forceinline__ unsigned hash32(unsigned x) {
    x ^= x >> 16; x *= 0x7feb352dU;
    x ^= x >> 15; x *= 0x846ca68bU;
    x ^= x >> 16;
    return x;
}

__global__ __launch_bounds__(256)
void fattn_bf16_wmma(const float* __restrict__ Q,
                     const float* __restrict__ KV,
                     float* __restrict__ Out)
{
    __shared__ __align__(16) __bf16 ldsK [CHUNK * DHEAD];                 // [kv][d]
    __shared__ __align__(16) __bf16 ldsKt[DHEAD * CHUNK];                 // [d][kv]
    __shared__ __align__(16) __bf16 ldsP [WAVES][ROWS_PER_WAVE * CHUNK];  // per-wave P

    const int tid    = threadIdx.x;
    const int wave   = tid >> 5;
    const int lane   = tid & 31;
    const int hlf    = lane >> 4;      // wave32 half
    const int lane16 = lane & 15;

    const int rowBlock = blockIdx.x * ROWS_PER_BLOCK;     // flattened (b*SEQ + q)
    const int b        = rowBlock / SEQ;
    const int qbase    = (rowBlock % SEQ) + wave * ROWS_PER_WAVE;
    const unsigned qflat = (unsigned)(b * SEQ + qbase);

    // ---- Q A-fragments (prescaled into exp2 domain): two 16x32 bf16 tiles ----
    V16U qa[2];
    {
        const float* qp = Q + ((long long)b * SEQ + (qbase + lane16)) * DHEAD;
        #pragma unroll
        for (int c = 0; c < 2; ++c) {
            float t[16];
            const float* p0 = qp + c * 32 + hlf * 8;
            const float* p1 = qp + c * 32 + 16 + hlf * 8;
            *(float4*)&t[0]  = *(const float4*)(p0);
            *(float4*)&t[4]  = *(const float4*)(p0 + 4);
            *(float4*)&t[8]  = *(const float4*)(p1);
            *(float4*)&t[12] = *(const float4*)(p1 + 4);
            #pragma unroll
            for (int j = 0; j < 16; ++j) qa[c].v[j] = (__bf16)(t[j] * QSCALE);
        }
    }

    v8f o[4] = {};
    float mrow[8], lrow[8];
    #pragma unroll
    for (int r = 0; r < 8; ++r) { mrow[r] = -__builtin_inff(); lrow[r] = 0.0f; }

    for (int kb = 0; kb < SEQ; kb += CHUNK) {
        __syncthreads();
        // ---- stage 64x64 f32 KV chunk -> bf16, row-major + transposed -------
        {
            const float* src = KV + ((long long)b * SEQ + kb) * DHEAD;
            const int e = tid * 16;          // 256 threads x 16 = 4096 elems
            const int r = e >> 6;            // kv row 0..63
            const int c = e & 63;            // d col: 0/16/32/48
            float t[16];
            *(float4*)&t[0]  = *(const float4*)(src + r * DHEAD + c);
            *(float4*)&t[4]  = *(const float4*)(src + r * DHEAD + c + 4);
            *(float4*)&t[8]  = *(const float4*)(src + r * DHEAD + c + 8);
            *(float4*)&t[12] = *(const float4*)(src + r * DHEAD + c + 12);
            V8U w0, w1;
            #pragma unroll
            for (int j = 0; j < 8; ++j) { w0.e[j] = (__bf16)t[j]; w1.e[j] = (__bf16)t[8 + j]; }
            *(v8bf*)&ldsK[r * DHEAD + c]     = w0.v;
            *(v8bf*)&ldsK[r * DHEAD + c + 8] = w1.v;
            #pragma unroll
            for (int j = 0; j < 8; ++j) {
                ldsKt[(c + j)     * CHUNK + r] = w0.e[j];
                ldsKt[(c + j + 8) * CHUNK + r] = w1.e[j];
            }
            // CDNA5 global_prefetch_b8 of next chunk (L2-bound stream hint)
            if (kb + CHUNK < SEQ)
                __builtin_prefetch(src + CHUNK * DHEAD + r * DHEAD + c, 0, 0);
        }
        __syncthreads();

        // ---- S' = (Q*scale) K^T : four 16x16 kv tiles, K-dim 64 = 2 x k32 ---
        v8f s[4] = {};
        #pragma unroll
        for (int t = 0; t < 4; ++t) {
            #pragma unroll
            for (int c = 0; c < 2; ++c) {
                V16U bk;   // B 32x16: lane col = kv row, K = d in table order
                const __bf16* base = &ldsK[(t * 16 + lane16) * DHEAD + c * 32 + hlf * 8];
                bk.h[0] = *(const v8bf*)(base);
                bk.h[1] = *(const v8bf*)(base + 16);
                s[t] = __builtin_amdgcn_wmma_f32_16x16x32_bf16(false, qa[c].v, false, bk.v,
                                                               (short)0, s[t], false, false);
            }
        }

        // ---- online softmax (exp2 domain) + hash dropout; P -> LDS ----------
        #pragma unroll
        for (int r = 0; r < 8; ++r) {
            const float s0 = s[0][r], s1 = s[1][r], s2 = s[2][r], s3 = s[3][r];
            float cm = fmaxf(fmaxf(s0, s1), fmaxf(s2, s3));
            cm = redmax16(cm);
            const float mnew = fmaxf(mrow[r], cm);
            const float a  = exp2f(mrow[r] - mnew);
            const float e0 = exp2f(s0 - mnew);
            const float e1 = exp2f(s1 - mnew);
            const float e2 = exp2f(s2 - mnew);
            const float e3 = exp2f(s3 - mnew);
            lrow[r] = lrow[r] * a + redsum16((e0 + e1) + (e2 + e3));
            mrow[r] = mnew;

            // one hash -> 4 byte-wide keep decisions (keep 51/256 ~ 0.2, x5)
            const unsigned rowg = qflat + (unsigned)(r + 8 * hlf);
            const unsigned hh   = hash32(rowg * 1024u + (unsigned)((kb >> 6) * 16 + lane16)
                                         ^ 0x9E3779B9u);
            const float d0 = (( hh        & 0xFFu) < 51u) ? e0 * 5.0f : 0.0f;
            const float d1 = (((hh >>  8) & 0xFFu) < 51u) ? e1 * 5.0f : 0.0f;
            const float d2 = (((hh >> 16) & 0xFFu) < 51u) ? e2 * 5.0f : 0.0f;
            const float d3 = (((hh >> 24) & 0xFFu) < 51u) ? e3 * 5.0f : 0.0f;

            const int row = r + 8 * hlf;
            __bf16* pr = &ldsP[wave][row * CHUNK];
            pr[lane16]      = (__bf16)d0;
            pr[lane16 + 16] = (__bf16)d1;
            pr[lane16 + 32] = (__bf16)d2;
            pr[lane16 + 48] = (__bf16)d3;

            o[0][r] *= a; o[1][r] *= a; o[2][r] *= a; o[3][r] *= a;
        }

        // wave-private LDS round trip: DS ops are in-order per wave
        asm volatile("s_wait_dscnt 0x0" ::: "memory");

        // ---- O += P V : two K=32 chunks x four 16-wide d tiles --------------
        #pragma unroll
        for (int c = 0; c < 2; ++c) {
            V16U pa;   // A 16x32 of P for kv chunk c
            const __bf16* pb = &ldsP[wave][0] + lane16 * CHUNK + c * 32 + hlf * 8;
            pa.h[0] = *(const v8bf*)(pb);
            pa.h[1] = *(const v8bf*)(pb + 16);
            #pragma unroll
            for (int t = 0; t < 4; ++t) {
                V16U bv;   // B 32x16 from transposed LDS: lane col = d, K = kv
                const __bf16* base = &ldsKt[(t * 16 + lane16) * CHUNK + c * 32 + hlf * 8];
                bv.h[0] = *(const v8bf*)(base);
                bv.h[1] = *(const v8bf*)(base + 16);
                o[t] = __builtin_amdgcn_wmma_f32_16x16x32_bf16(false, pa.v, false, bv.v,
                                                               (short)0, o[t], false, false);
            }
        }
    }

    // ---- normalize and store ------------------------------------------------
    float* outp = Out + ((long long)b * SEQ + qbase) * DHEAD;
    #pragma unroll
    for (int r = 0; r < 8; ++r) {
        const float inv = 1.0f / lrow[r];
        float* op = outp + (r + 8 * hlf) * DHEAD + lane16;
        op[0]  = o[0][r] * inv;
        op[16] = o[1][r] * inv;
        op[32] = o[2][r] * inv;
        op[48] = o[3][r] * inv;
    }
}

extern "C" void kernel_launch(void* const* d_in, const int* in_sizes, int n_in,
                              void* d_out, int out_size, void* d_ws, size_t ws_size,
                              hipStream_t stream) {
    (void)in_sizes; (void)n_in; (void)out_size; (void)d_ws; (void)ws_size;
    const float* x3 = (const float*)d_in[0];   // Q   [4, 4096, 64]
    const float* x4 = (const float*)d_in[1];   // K=V [4, 4096, 64]
    float* out = (float*)d_out;                // [4, 4096, 64]
    dim3 grid((4 * SEQ) / ROWS_PER_BLOCK);     // 128 blocks
    dim3 block(256);                            // 8 wave32
    hipLaunchKernelGGL(fattn_bf16_wmma, grid, block, 0, stream, x3, x4, out);
}
